// OptionCriticGNN_67662914781648
// MI455X (gfx1250) — compile-verified, matrix operand order checked by reading, not run on hardware
//
#include <hip/hip_runtime.h>
#include <hip/hip_bf16.h>
#include <math.h>

typedef __attribute__((ext_vector_type(16))) __bf16 v16bf;
typedef __attribute__((ext_vector_type(8)))  float  v8f;

#define NPG 5
#define HEADS 8
#define HID 64
#define DIMF 512
#define GRAPHS 8192
#define NNODES (GRAPHS*NPG)

typedef unsigned short bf16r;   // raw bf16 bits

__device__ __forceinline__ bf16r f2bf(float f) {
  unsigned int u = __float_as_uint(f);
  u += 0x7FFFu + ((u >> 16) & 1u);
  return (bf16r)(u >> 16);
}
__device__ __forceinline__ float bf2f(bf16r h) {
  return __uint_as_float((unsigned int)h << 16);
}

// ---- CDNA5 async global->LDS copy (ASYNCcnt path), with safe fallback ----
#if defined(__has_builtin)
#if __has_builtin(__builtin_amdgcn_global_load_async_to_lds_b64) && \
    __has_builtin(__builtin_amdgcn_s_wait_asynccnt)
#define USE_ASYNC 1
#endif
#endif
#ifndef USE_ASYNC
#define USE_ASYNC 0
#endif

typedef int v2i __attribute__((vector_size(8)));   // <2 x i32>, matches builtin
typedef __attribute__((address_space(1))) v2i gv2i;
typedef __attribute__((address_space(3))) v2i sv2i;

__device__ __forceinline__ void cp8(void* lds, const void* g) {
#if USE_ASYNC
  __builtin_amdgcn_global_load_async_to_lds_b64((gv2i*)g, (sv2i*)lds, 0, 0);
#else
  *(unsigned long long*)lds = *(const unsigned long long*)g;
#endif
}
__device__ __forceinline__ void cp_wait() {
#if USE_ASYNC
  __builtin_amdgcn_s_wait_asynccnt(0);
#endif
}

union Frag { v16bf v; unsigned int u[8]; };

#define LSTR 40   // LDS row stride in bf16 (80B = 20 dwords -> conflict-free gathers)

// C[M,N](bf16) = act(A[M,K](bf16) @ Bt^T + bias). Bt is pre-transposed [N,K] bf16.
// Columns k >= splitK come from A2[row/rowdiv] (actor concat); splitK % 32 == 0 so
// the source select is uniform per K-step.  M%128==0, N%64==0, K%32==0: no guards.
__global__ __launch_bounds__(256)
void gemm_bf16_wmma(const bf16r* __restrict__ A, int strideA,
                    const bf16r* __restrict__ A2, int strideA2,
                    int splitK, int rowdiv,
                    const bf16r* __restrict__ Bt,
                    const float* __restrict__ bias,
                    bf16r* __restrict__ C,
                    int M, int N, int K, int relu)
{
  __shared__ bf16r Ash[128 * LSTR];
  __shared__ bf16r Bsh[64 * LSTR];

  const int tid  = threadIdx.x;
  const int wv   = tid >> 5;
  const int lane = tid & 31;
  const int half = lane >> 4;
  const int l15  = lane & 15;
  const int m0   = blockIdx.y * 128;
  const int n0   = blockIdx.x * 64;

  v8f acc[4] = {};

  for (int k0 = 0; k0 < K; k0 += 32) {
    __syncthreads();
    const bool second = (k0 >= splitK);                 // uniform
    const bf16r* Abase = second ? A2 : A;
    const int    astr  = second ? strideA2 : strideA;
    const int    kk    = second ? (k0 - splitK) : k0;

    #pragma unroll
    for (int i = 0; i < 4; ++i) {                       // A tile: 128x32, 8B chunks
      int q = i * 256 + tid;
      int r = q >> 3, kc = (q & 7) * 4;
      int rowg = m0 + r;
      int arow = second ? (rowg / rowdiv) : rowg;
      cp8(Ash + r * LSTR + kc, Abase + (size_t)arow * astr + kk + kc);
    }
    #pragma unroll
    for (int i = 0; i < 2; ++i) {                       // B^T tile: 64x32
      int q = i * 256 + tid;
      int n = q >> 3, kc = (q & 7) * 4;
      cp8(Bsh + n * LSTR + kc, Bt + (size_t)(n0 + n) * K + k0 + kc);
    }
    cp_wait();
    __syncthreads();

    const int kb2 = half * 4;                           // ISA 16-bit 16x32 layout
    Frag af;
    const unsigned int* Arow = (const unsigned int*)(Ash + (wv * 16 + l15) * LSTR);
    #pragma unroll
    for (int i = 0; i < 4; ++i) { af.u[i] = Arow[kb2 + i]; af.u[4 + i] = Arow[kb2 + 8 + i]; }

    #pragma unroll
    for (int t = 0; t < 4; ++t) {
      Frag bfv;
      const unsigned int* Brow = (const unsigned int*)(Bsh + (t * 16 + l15) * LSTR);
      #pragma unroll
      for (int i = 0; i < 4; ++i) { bfv.u[i] = Brow[kb2 + i]; bfv.u[4 + i] = Brow[kb2 + 8 + i]; }
      acc[t] = __builtin_amdgcn_wmma_f32_16x16x32_bf16(
          false, af.v, false, bfv.v, (short)0, acc[t], false, false);
    }
  }

  #pragma unroll
  for (int t = 0; t < 4; ++t) {                         // D: VGPR r -> M=r+8*half
    int n = n0 + t * 16 + l15;
    float bv = bias ? bias[n] : 0.f;
    #pragma unroll
    for (int r = 0; r < 8; ++r) {
      int m = m0 + wv * 16 + half * 8 + r;
      float v = acc[t][r] + bv;
      if (relu) v = fmaxf(v, 0.f);
      C[(size_t)m * N + n] = f2bf(v);
    }
  }
}

// ---- conversion / pre-transpose passes (tiny, run once per call) ----
__global__ void cvt_x(const float* __restrict__ x, bf16r* __restrict__ xb)
{ // [N,29] f32 -> [N,32] bf16 zero-padded
  int t = blockIdx.x * blockDim.x + threadIdx.x;
  if (t >= NNODES * 32) return;
  int r = t >> 5, k = t & 31;
  xb[t] = (k < 29) ? f2bf(x[r * 29 + k]) : (bf16r)0;
}

__global__ void cvt_wT(const float* __restrict__ W, bf16r* __restrict__ Wt,
                       int K, int N, int Kp)
{ // [K,N] f32 -> [N,Kp] bf16 transposed, zero-padded
  int t = blockIdx.x * blockDim.x + threadIdx.x;
  if (t >= N * Kp) return;
  int n = t / Kp, k = t % Kp;
  Wt[t] = (k < K) ? f2bf(W[k * N + n]) : (bf16r)0;
}

// One wave per 5-node fully-connected graph (+self loops). bf16 in/out, f32 math.
__global__ __launch_bounds__(128)
void gat_layer(const bf16r* __restrict__ hlin, const float* __restrict__ eattr,
               const float* __restrict__ a_s, const float* __restrict__ a_d,
               const float* __restrict__ We, const float* __restrict__ a_e,
               const float* __restrict__ bias, bf16r* __restrict__ out)
{
  __shared__ float hl[4][NPG * DIMF];
  __shared__ float ss[4][NPG * HEADS];
  __shared__ float dd[4][NPG * HEADS];
  __shared__ float lg[4][NPG * HEADS * NPG];
  __shared__ float wdot[4][HEADS];

  const int w = threadIdx.x >> 5, lane = threadIdx.x & 31;
  const int g = blockIdx.x * 4 + w;
  const size_t base = (size_t)g * NPG * DIMF;

  for (int i = lane; i < NPG * DIMF; i += 32) hl[w][i] = bf2f(hlin[base + i]);
  if (lane < HEADS) {
    float acc = 0.f;
    for (int c = 0; c < HID; ++c) acc += We[lane * HID + c] * a_e[lane * HID + c];
    wdot[w][lane] = acc;
  }
  __syncthreads();

  for (int p = lane; p < NPG * HEADS; p += 32) {
    int n = p / HEADS, hd = p % HEADS;
    const float* hr = &hl[w][n * DIMF + hd * HID];
    float sa = 0.f, da = 0.f;
    for (int c = 0; c < HID; ++c) {
      float hv = hr[c];
      sa += hv * a_s[hd * HID + c];
      da += hv * a_d[hd * HID + c];
    }
    ss[w][p] = sa; dd[w][p] = da;
  }
  __syncthreads();

  for (int p = lane; p < NPG * HEADS * NPG; p += 32) {
    int dst = p / (HEADS * NPG);
    int hd  = (p / NPG) % HEADS;
    int src = p % NPG;
    float ev = 0.f;
    if (src != dst) {
      int kk = src * 4 + (dst < src ? dst : dst - 1);
      ev = eattr[(size_t)g * 20 + kk];
    }
    float l = ss[w][src * HEADS + hd] + dd[w][dst * HEADS + hd] + ev * wdot[w][hd];
    lg[w][p] = (l > 0.f) ? l : 0.2f * l;
  }
  __syncthreads();

  for (int p = lane; p < NPG * HEADS; p += 32) {
    int dst = p / HEADS, hd = p % HEADS;
    float* row = &lg[w][dst * (HEADS * NPG) + hd * NPG];
    float m = row[0];
    for (int s = 1; s < NPG; ++s) m = fmaxf(m, row[s]);
    float e[NPG], sum = 0.f;
    for (int s = 0; s < NPG; ++s) { e[s] = __expf(row[s] - m); sum += e[s]; }
    float inv = 1.f / (sum + 1e-16f);
    for (int s = 0; s < NPG; ++s) row[s] = e[s] * inv;
  }
  __syncthreads();

  for (int dst = 0; dst < NPG; ++dst) {
    for (int c = lane; c < DIMF; c += 32) {
      int hd = c >> 6;
      const float* al = &lg[w][dst * (HEADS * NPG) + hd * NPG];
      float acc = bias[c];
      for (int s = 0; s < NPG; ++s) acc += al[s] * hl[w][s * DIMF + c];
      out[base + dst * DIMF + c] = f2bf(fmaxf(acc, 0.f));
    }
  }
}

__global__ void mean_pool(const bf16r* __restrict__ h, bf16r* __restrict__ g)
{
  int t = blockIdx.x * blockDim.x + threadIdx.x;
  if (t >= GRAPHS * DIMF) return;
  int gi = t / DIMF, c = t % DIMF;
  size_t b = (size_t)gi * NPG * DIMF + c;
  float s = 0.f;
  for (int a = 0; a < NPG; ++a) s += bf2f(h[b + (size_t)a * DIMF]);
  g[t] = f2bf(s * 0.2f);
}

__global__ void head2(const bf16r* __restrict__ H, const float* __restrict__ w2,
                      const float* __restrict__ b2, float* __restrict__ out,
                      int Kh, int No, int colOff, int sig)
{
  int t = blockIdx.x * blockDim.x + threadIdx.x;
  if (t >= GRAPHS * No) return;
  int gi = t / No, j = t % No;
  const bf16r* hr = &H[(size_t)gi * Kh];
  float acc = b2[j];
  for (int k = 0; k < Kh; ++k) acc += bf2f(hr[k]) * w2[k * No + j];
  if (sig) acc = 1.f / (1.f + __expf(-acc));
  out[(size_t)gi * 39 + colOff + j] = acc;
}

__global__ void actor_final(const bf16r* __restrict__ Hh, const float* __restrict__ aw2,
                            const float* __restrict__ ab2, const float* __restrict__ lim,
                            float* __restrict__ out)
{
  int t = blockIdx.x * blockDim.x + threadIdx.x;
  if (t >= NNODES * 6) return;
  int node = t / 6, j = t % 6;
  int gi = node / NPG, a = node % NPG;
  const bf16r* hr = &Hh[(size_t)node * 128];
  float acc = ab2[j];
  for (int k = 0; k < 128; ++k) acc += bf2f(hr[k]) * aw2[k * 6 + j];
  if (j < 3) {
    out[(size_t)gi * 39 + 9 + a * 3 + j] = tanhf(acc) * lim[j];
  } else {
    float sg = 1.f / (1.f + __expf(-acc));
    out[(size_t)gi * 39 + 24 + a * 3 + (j - 3)] = 0.05f + sg * 0.45f + 1e-6f;
  }
}

extern "C" void kernel_launch(void* const* d_in, const int* in_sizes, int n_in,
                              void* d_out, int out_size, void* d_ws, size_t ws_size,
                              hipStream_t stream)
{
  (void)in_sizes; (void)n_in; (void)out_size; (void)ws_size;
  const float* x     = (const float*)d_in[0];
  const float* eattr = (const float*)d_in[1];
  const float* W1  = (const float*)d_in[5];
  const float* as1 = (const float*)d_in[6];
  const float* ad1 = (const float*)d_in[7];
  const float* We1 = (const float*)d_in[8];
  const float* ae1 = (const float*)d_in[9];
  const float* b1  = (const float*)d_in[10];
  const float* W2  = (const float*)d_in[11];
  const float* as2 = (const float*)d_in[12];
  const float* ad2 = (const float*)d_in[13];
  const float* We2 = (const float*)d_in[14];
  const float* ae2 = (const float*)d_in[15];
  const float* b2  = (const float*)d_in[16];
  const float* pw1 = (const float*)d_in[17];
  const float* pb1 = (const float*)d_in[18];
  const float* pw2 = (const float*)d_in[19];
  const float* pb2 = (const float*)d_in[20];
  const float* qw1 = (const float*)d_in[21];
  const float* qb1 = (const float*)d_in[22];
  const float* qw2 = (const float*)d_in[23];
  const float* qb2 = (const float*)d_in[24];
  const float* aw1 = (const float*)d_in[25];
  const float* ab1 = (const float*)d_in[26];
  const float* aw2 = (const float*)d_in[27];
  const float* ab2 = (const float*)d_in[28];
  const float* tw1 = (const float*)d_in[29];
  const float* tb1 = (const float*)d_in[30];
  const float* tw2 = (const float*)d_in[31];
  const float* tb2 = (const float*)d_in[32];
  const float* lim = (const float*)d_in[33];
  float* out = (float*)d_out;

  bf16r* p = (bf16r*)d_ws;
  bf16r* xbf   = p; p += (size_t)NNODES * 32;
  bf16r* h1lin = p; p += (size_t)NNODES * DIMF;
  bf16r* h1    = p; p += (size_t)NNODES * DIMF;
  bf16r* gbf   = p; p += (size_t)GRAPHS * DIMF;
  bf16r* pih   = p; p += (size_t)GRAPHS * 128;
  bf16r* qih   = p; p += (size_t)GRAPHS * 128;
  bf16r* tih   = p; p += (size_t)GRAPHS * 64;
  bf16r* acth  = p; p += (size_t)NNODES * 128;
  bf16r* w1t   = p; p += (size_t)512 * 32;
  bf16r* w2t   = p; p += (size_t)512 * 512;
  bf16r* p1t   = p; p += (size_t)128 * 512;
  bf16r* q1t   = p; p += (size_t)128 * 512;
  bf16r* t1t   = p; p += (size_t)64 * 512;
  bf16r* a1t   = p; p += (size_t)128 * 1024;
  bf16r* h2lin = h1lin;   // reuse
  bf16r* h2    = h1;      // reuse

  auto cvtw = [&](const float* W, bf16r* Wt, int K, int N, int Kp) {
    cvt_wT<<<(N * Kp + 255) / 256, 256, 0, stream>>>(W, Wt, K, N, Kp);
  };
  auto gemm = [&](const bf16r* A, int sA, const bf16r* A2, int sA2, int splitK,
                  int rdiv, const bf16r* Bt, const float* bias, bf16r* C,
                  int M, int Nn, int K, int relu) {
    dim3 grid(Nn / 64, M / 128);
    gemm_bf16_wmma<<<grid, 256, 0, stream>>>(A, sA, A2, sA2, splitK, rdiv,
                                             Bt, bias, C, M, Nn, K, relu);
  };

  // pre-convert activations & weights to bf16 (weights pre-transposed [N,Kp])
  cvt_x<<<(NNODES * 32 + 255) / 256, 256, 0, stream>>>(x, xbf);
  cvtw(W1, w1t, 29, 512, 32);
  cvtw(W2, w2t, 512, 512, 512);
  cvtw(pw1, p1t, 512, 128, 512);
  cvtw(qw1, q1t, 512, 128, 512);
  cvtw(tw1, t1t, 512, 64, 512);
  cvtw(aw1, a1t, 1024, 128, 1024);

  // conv1 linear (K padded to 32)
  gemm(xbf, 32, nullptr, 0, 32, 1, w1t, nullptr, h1lin, NNODES, DIMF, 32, 0);
  gat_layer<<<GRAPHS / 4, 128, 0, stream>>>(h1lin, eattr, as1, ad1, We1, ae1, b1, h1);
  // conv2 linear (dominant 40960x512x512)
  gemm(h1, DIMF, nullptr, 0, DIMF, 1, w2t, nullptr, h2lin, NNODES, DIMF, DIMF, 0);
  gat_layer<<<GRAPHS / 4, 128, 0, stream>>>(h2lin, eattr, as2, ad2, We2, ae2, b2, h2);
  mean_pool<<<(GRAPHS * DIMF + 255) / 256, 256, 0, stream>>>(h2, gbf);
  // head first layers
  gemm(gbf, DIMF, nullptr, 0, DIMF, 1, p1t, pb1, pih, GRAPHS, 128, DIMF, 1);
  gemm(gbf, DIMF, nullptr, 0, DIMF, 1, q1t, qb1, qih, GRAPHS, 128, DIMF, 1);
  gemm(gbf, DIMF, nullptr, 0, DIMF, 1, t1t, tb1, tih, GRAPHS, 64, DIMF, 1);
  // actor first layer: virtual concat [h2 | g[row/5]], K=1024, split at 512
  gemm(h2, DIMF, gbf, DIMF, DIMF, NPG, a1t, ab1, acth, NNODES, 128, 1024, 1);
  // tiny second layers + output transforms
  head2<<<(GRAPHS * 4 + 255) / 256, 256, 0, stream>>>(pih, pw2, pb2, out, 128, 4, 0, 0);
  head2<<<(GRAPHS * 4 + 255) / 256, 256, 0, stream>>>(qih, qw2, qb2, out, 128, 4, 4, 0);
  head2<<<(GRAPHS + 255) / 256, 256, 0, stream>>>(tih, tw2, tb2, out, 64, 1, 8, 1);
  actor_final<<<(NNODES * 6 + 255) / 256, 256, 0, stream>>>(acth, aw2, ab2, lim, out);
}
